// PointTransformerCls_1400159339080
// MI455X (gfx1250) — matrix operand order repaired and checked by reading, not compiled, not run on hardware
//
#include <hip/hip_runtime.h>
#include <hip/hip_bf16.h>

// ================= CDNA5 WMMA types =================
typedef _Float16 half_t;
typedef __attribute__((ext_vector_type(16))) _Float16 v16h;
typedef __attribute__((ext_vector_type(8)))  float    v8f;
typedef __attribute__((ext_vector_type(4)))  float    v4f;

#define LN_EPS_F 1e-5f
#define BATCH 4

// ---------------------------------------------------------------------------
// Generic GEMM: Y[rows,Cout] = act(X[rows,Cin] @ W[Cin,Cout] + bias)
// f32 in memory, v_wmma_f32_16x16x32_f16, f32 accumulate.
// Whole W staged once per block into LDS, TRANSPOSED + f16 + zero-padded:
//   ldsWT[n * Cin_pad + k]   (Cin_pad = ceil32(Cin), Cout_pad = ceil16(Cout))
// -> B fragment = two aligned ds_load_b128 per K-step, no conditionals.
// A fragment: fast tiles (full 16 rows, full K step) use 4x global_load_b128;
// the fast/edge split is hoisted OUT of the K-loop so the hot loop is
// branch-free (no exec-mask toggles, loads pipeline in one clause).
//
// Layouts: A (ISA 16-bit A 16x32 table): lane l row M=(l&15),
//   elem e -> K = kbA + e + (e>=8?8:0), kbA=(l>=16)?8:0.
// B: lane l col N=(l&15), elem e -> K = kbB + e + (e>=8?8:0),
//   kbB=(l>=16)?8:0  -> two 8-half runs in ldsWT row (same order as A).
// C: lane l -> n=(l&15), vgpr r -> m=((l>=16)?8:0)+r.
// ---------------------------------------------------------------------------
__global__ void __launch_bounds__(256)
gemm_wmma_kernel(const float* __restrict__ X, const float* __restrict__ W,
                 const float* __restrict__ bias, float* __restrict__ Y,
                 long rows, int Cin, int Cout, int relu) {
  __shared__ _Float16 ldsWT[16384];  // 32 KB, fits 128x128 padded
  const int tid  = threadIdx.x;
  const int l    = tid & 31;
  const int wave = tid >> 5;
  const int Cin_pad  = (Cin + 31) & ~31;
  const int Cout_pad = (Cout + 15) & ~15;

  if (Cin != Cin_pad || Cout != Cout_pad) {      // block-uniform predicate
    for (int i = tid; i < Cout_pad * Cin_pad; i += 256) ldsWT[i] = (half_t)0.f;
    __syncthreads();
  }
  for (int i = tid; i < Cin * Cout; i += 256) {
    int k = i / Cout, n = i % Cout;              // coalesced global read
    ldsWT[n * Cin_pad + k] = (half_t)W[i];       // transposed LDS write
  }
  __syncthreads();

  const long tilesN = Cout_pad >> 4;
  const long tilesM = (rows + 15) >> 4;
  const long tile   = (long)blockIdx.x * 8 + wave;
  if (tile >= tilesM * tilesN) return;           // wave-uniform
  const long m0 = (tile / tilesN) * 16;
  const int  n0 = (int)(tile % tilesN) * 16;

  const int  kbA  = (l >= 16) ? 8 : 0;
  const int  kbB  = (l >= 16) ? 8 : 0;
  typedef __attribute__((ext_vector_type(8))) _Float16 v8h_;
  auto ldB = [](const _Float16* p) -> v16h {
    v8h_ lo = *(const v8h_*)p, hi = *(const v8h_*)(p + 16);
    return __builtin_shufflevector(lo, hi, 0, 1, 2, 3, 4, 5, 6, 7, 8, 9, 10, 11, 12, 13, 14, 15);
  };
  const long rowA = m0 + (l & 15);
  const int  colB = n0 + (l & 15);               // < Cout_pad always
  const _Float16* wrow = ldsWT + (long)colB * Cin_pad;
  const bool rowsFull = (m0 + 16) <= rows;       // wave-uniform

  v8f c;
#pragma unroll
  for (int r = 0; r < 8; ++r) c[r] = 0.f;

  if (rowsFull) {
    // ---- branch-free hot loop over full K steps ----
    const int kFull = Cin & ~31;
    const float* xrow = X + rowA * (long)Cin;
    for (int k0 = 0; k0 < kFull; k0 += 32) {
      const v4f* pa = (const v4f*)(xrow + k0 + kbA);
      __builtin_prefetch(xrow + k0 + 32, 0, 0);
      v4f f0 = pa[0], f1 = pa[1], f2 = pa[4], f3 = pa[5];
      v16h a;
#pragma unroll
      for (int e = 0; e < 4; ++e) {
        a[e]      = (half_t)f0[e];
        a[4 + e]  = (half_t)f1[e];
        a[8 + e]  = (half_t)f2[e];
        a[12 + e] = (half_t)f3[e];
      }
      v16h b = ldB(wrow + k0 + kbB);             // 2x ds_load_b128
      c = __builtin_amdgcn_wmma_f32_16x16x32_f16(false, a, false, b, (short)0, c,
                                                 false, false);
    }
    if (kFull < Cin) {                           // K tail (Cin = 3 or 12 only)
      v16h a;
#pragma unroll
      for (int e = 0; e < 16; ++e) {
        int ka  = kFull + kbA + e + ((e >= 8) ? 8 : 0);
        int kaC = (ka < Cin) ? ka : 0;
        float xv = xrow[kaC];                    // unconditional clamped load
        a[e] = (half_t)((ka < Cin) ? xv : 0.f);
      }
      v16h b = ldB(wrow + kFull + kbB);
      c = __builtin_amdgcn_wmma_f32_16x16x32_f16(false, a, false, b, (short)0, c,
                                                 false, false);
    }
  } else {
    // ---- edge tiles (partial rows; only the 4-row head GEMMs) ----
    long rowC = (rowA < rows) ? rowA : (rows - 1);
    const float* xr = X + rowC * (long)Cin;
    for (int k0 = 0; k0 < Cin; k0 += 32) {
      v16h a;
#pragma unroll
      for (int e = 0; e < 16; ++e) {
        int ka  = k0 + kbA + e + ((e >= 8) ? 8 : 0);
        int kaC = (ka < Cin) ? ka : 0;
        float xv = xr[kaC];                      // unconditional clamped load
        a[e] = (half_t)(((rowA < rows) && (ka < Cin)) ? xv : 0.f);
      }
      v16h b = ldB(wrow + k0 + kbB);
      c = __builtin_amdgcn_wmma_f32_16x16x32_f16(false, a, false, b, (short)0, c,
                                                 false, false);
    }
  }

  const int mb = (l >= 16) ? 8 : 0;
  if (colB < Cout) {
    float bv = bias ? bias[colB] : 0.f;
    if (rowsFull) {
#pragma unroll
      for (int r = 0; r < 8; ++r) {
        float y = c[r] + bv;
        if (relu) y = fmaxf(y, 0.f);
        Y[(m0 + mb + r) * (long)Cout + colB] = y;
      }
    } else {
#pragma unroll
      for (int r = 0; r < 8; ++r) {
        long m = m0 + mb + r;
        if (m < rows) {
          float y = c[r] + bv;
          if (relu) y = fmaxf(y, 0.f);
          Y[m * (long)Cout + colB] = y;
        }
      }
    }
  }
}

// ---------------------------------------------------------------------------
// Fused kNN: WMMA 16x16 distance tiles + in-register top-16 (ascending).
// Rank key s = |r|^2 - 2 q.r (|q|^2 constant per query -> order-invariant).
// ---------------------------------------------------------------------------
__device__ __forceinline__ void topk_insert(float (&d)[16], int (&ix)[16],
                                            float nd, int ni) {
  if (nd < d[15]) {
    d[15] = nd; ix[15] = ni;
#pragma unroll
    for (int j = 15; j > 0; --j) {
      if (d[j] < d[j - 1]) {
        float tf = d[j]; d[j] = d[j - 1]; d[j - 1] = tf;
        int   ti = ix[j]; ix[j] = ix[j - 1]; ix[j - 1] = ti;
      }
    }
  }
}

__global__ void __launch_bounds__(128)
knn_kernel(const float* __restrict__ qpos, const float* __restrict__ rpos,
           int* __restrict__ idx_out, int Mq, int Nr, int blocksPerBatch) {
  __shared__ float sd[4][32][32];
  __shared__ int   si[4][32][32];
  const int l    = threadIdx.x & 31;
  const int wv   = threadIdx.x >> 5;
  const int b    = blockIdx.x / blocksPerBatch;
  const int qblk = blockIdx.x % blocksPerBatch;
  const int q0   = qblk * 128 + wv * 32;
  const float* qp = qpos + (long)b * Mq * 3;
  const float* rp = rpos + (long)b * Nr * 3;

  v16h bq1, bq2;
#pragma unroll
  for (int e = 0; e < 16; ++e) { bq1[e] = (half_t)0.f; bq2[e] = (half_t)0.f; }
  if (l < 16) {   // lanes 0..15 carry K=0..2; lanes 16..31 hold K>=8 -> zero
    int qa = q0 + l, qb = q0 + 16 + l;
    bq1[0] = (half_t)qp[qa * 3 + 0]; bq1[1] = (half_t)qp[qa * 3 + 1];
    bq1[2] = (half_t)qp[qa * 3 + 2];
    bq2[0] = (half_t)qp[qb * 3 + 0]; bq2[1] = (half_t)qp[qb * 3 + 1];
    bq2[2] = (half_t)qp[qb * 3 + 2];
  }

  float da[16], db_[16]; int ia[16], ib_[16];
#pragma unroll
  for (int j = 0; j < 16; ++j) { da[j] = 3.0e38f; db_[j] = 3.0e38f; ia[j] = 0; ib_[j] = 0; }

  const int mbase  = (l >= 16) ? 8 : 0;
  const int ntiles = Nr >> 4;
  for (int rt = 0; rt < ntiles; ++rt) {
    int rr = rt * 16 + (l & 15);
    float rx = rp[rr * 3 + 0], ry = rp[rr * 3 + 1], rz = rp[rr * 3 + 2];
    float rn = rx * rx + ry * ry + rz * rz;
    v16h ar;
#pragma unroll
    for (int e = 0; e < 16; ++e) ar[e] = (half_t)0.f;
    if (l < 16) { ar[0] = (half_t)rx; ar[1] = (half_t)ry; ar[2] = (half_t)rz; }
    v8f zc;
#pragma unroll
    for (int r = 0; r < 8; ++r) zc[r] = 0.f;
    v8f c1 = __builtin_amdgcn_wmma_f32_16x16x32_f16(false, ar, false, bq1,
                                                    (short)0, zc, false, false);
    v8f c2 = __builtin_amdgcn_wmma_f32_16x16x32_f16(false, ar, false, bq2,
                                                    (short)0, zc, false, false);
#pragma unroll
    for (int r = 0; r < 8; ++r) {
      int m = mbase + r;
      float rnm = __shfl(rn, m, 32);
      int ridx = rt * 16 + m;
      topk_insert(da,  ia,  rnm - 2.f * c1[r], ridx);
      topk_insert(db_, ib_, rnm - 2.f * c2[r], ridx);
    }
  }

  const int h = (l >= 16) ? 16 : 0;
#pragma unroll
  for (int j = 0; j < 16; ++j) {
    sd[wv][l & 15][h + j] = da[j];          si[wv][l & 15][h + j] = ia[j];
    sd[wv][16 + (l & 15)][h + j] = db_[j];  si[wv][16 + (l & 15)][h + j] = ib_[j];
  }
  __syncthreads();
  int p = 0, q = 16;
  long ob = ((long)b * Mq + (q0 + l)) * 16;
  for (int t = 0; t < 16; ++t) {
    float dp = (p < 16) ? sd[wv][l][p] : 3.0e38f;
    float dq = (q < 32) ? sd[wv][l][q] : 3.0e38f;
    bool tp = dp <= dq;
    idx_out[ob + t] = tp ? si[wv][l][p] : si[wv][l][q];
    p += tp ? 1 : 0; q += tp ? 0 : 1;
  }
}

// ------------------ LayerNorm (+relu): one wave per row ------------------
__global__ void __launch_bounds__(256)
ln_kernel(const float* __restrict__ X, const float* __restrict__ g,
          const float* __restrict__ bta, float* __restrict__ Y,
          long rows, int C, int relu) {
  const int l = threadIdx.x & 31;
  long row = ((long)blockIdx.x * blockDim.x + threadIdx.x) >> 5;
  if (row >= rows) return;
  const float* x = X + row * (long)C;
  float s = 0.f, s2 = 0.f;
  for (int c = l; c < C; c += 32) { float v = x[c]; s += v; s2 += v * v; }
#pragma unroll
  for (int o = 16; o > 0; o >>= 1) {
    s  += __shfl_xor(s, o, 32);
    s2 += __shfl_xor(s2, o, 32);
  }
  float mean = s / C;
  float var  = s2 / C - mean * mean;
  float inv  = rsqrtf(var + LN_EPS_F);
  float* y = Y + row * (long)C;
  for (int c = l; c < C; c += 32) {
    float v = (x[c] - mean) * inv * g[c] + bta[c];
    if (relu) v = fmaxf(v, 0.f);
    y[c] = v;
  }
}

// ------------------- float4 elementwise / gather kernels -------------------
// All feature widths at these call sites are multiples of 4.
__global__ void gather_feat_kernel(const float* __restrict__ f, const int* __restrict__ idx,
                                   float* __restrict__ out, int Nsrc, int Mdst, int C) {
  const int C4 = C >> 2;
  const v4f* f4 = (const v4f*)f;
  v4f* o4 = (v4f*)out;
  long total = (long)BATCH * Mdst * 16 * C4;
  for (long i = (long)blockIdx.x * blockDim.x + threadIdx.x; i < total;
       i += (long)gridDim.x * blockDim.x) {
    int c4 = (int)(i % C4); long t = i / C4;
    int j = (int)(t % 16); long t2 = t / 16;
    int m = (int)(t2 % Mdst); int b = (int)(t2 / Mdst);
    int src = idx[((long)b * Mdst + m) * 16 + j];
    o4[i] = f4[((long)b * Nsrc + src) * C4 + c4];
  }
}

__global__ void rel_kernel(const float* __restrict__ pos, const int* __restrict__ idx,
                           float* __restrict__ out, int N) {
  long total = (long)BATCH * N * 16 * 3;
  for (long i = (long)blockIdx.x * blockDim.x + threadIdx.x; i < total;
       i += (long)gridDim.x * blockDim.x) {
    int c = (int)(i % 3); long t = i / 3;
    int j = (int)(t % 16); long t2 = t / 16;
    int n = (int)(t2 % N); int b = (int)(t2 / N);
    int src = idx[((long)b * N + n) * 16 + j];
    out[i] = pos[((long)b * N + n) * 3 + c] - pos[((long)b * N + src) * 3 + c];
  }
}

__global__ void wcombine_kernel(const float* __restrict__ q, const float* __restrict__ kk,
                                const float* __restrict__ pe, float* __restrict__ w,
                                int N, int C) {
  const int C4 = C >> 2;
  const v4f* q4 = (const v4f*)q; const v4f* k4 = (const v4f*)kk;
  const v4f* p4 = (const v4f*)pe; v4f* w4 = (v4f*)w;
  long total = (long)BATCH * N * 16 * C4;
  for (long i = (long)blockIdx.x * blockDim.x + threadIdx.x; i < total;
       i += (long)gridDim.x * blockDim.x) {
    int c4 = (int)(i % C4); long t = i / C4;
    long t2 = t / 16;
    int n = (int)(t2 % N); int b = (int)(t2 / N);
    w4[i] = q4[((long)b * N + n) * C4 + c4] - k4[i] + p4[i];
  }
}

// softmax over the 16-neighbor axis, then sum_k attn * (v + pe); 4 channels/thread
__global__ void attn_kernel(const float* __restrict__ w, const float* __restrict__ v,
                            const float* __restrict__ pe, float* __restrict__ out,
                            int N, int C) {
  const int C4 = C >> 2;
  const v4f* w4 = (const v4f*)w; const v4f* v4 = (const v4f*)v;
  const v4f* p4 = (const v4f*)pe; v4f* o4 = (v4f*)out;
  long total = (long)BATCH * N * C4;
  for (long i = (long)blockIdx.x * blockDim.x + threadIdx.x; i < total;
       i += (long)gridDim.x * blockDim.x) {
    int c4 = (int)(i % C4); long t = i / C4;
    int n = (int)(t % N); int b = (int)(t / N);
    long base = ((long)b * N + n) * 16 * C4 + c4;
    v4f wr[16];
#pragma unroll
    for (int j = 0; j < 16; ++j) wr[j] = w4[base + (long)j * C4];
    v4f mx = wr[0];
#pragma unroll
    for (int j = 1; j < 16; ++j)
#pragma unroll
      for (int e = 0; e < 4; ++e) mx[e] = fmaxf(mx[e], wr[j][e]);
    v4f es, acc;
#pragma unroll
    for (int e = 0; e < 4; ++e) { es[e] = 0.f; acc[e] = 0.f; }
#pragma unroll
    for (int j = 0; j < 16; ++j) {
      v4f vp = v4[base + (long)j * C4] + p4[base + (long)j * C4];
#pragma unroll
      for (int e = 0; e < 4; ++e) {
        float ex = __expf(wr[j][e] - mx[e]);
        es[e]  += ex;
        acc[e] += ex * vp[e];
      }
    }
    v4f r;
#pragma unroll
    for (int e = 0; e < 4; ++e) r[e] = acc[e] / es[e];
    o4[i] = r;
  }
}

__global__ void relu_add_kernel(const float* __restrict__ a, const float* __restrict__ b,
                                float* __restrict__ y, long n4) {
  const v4f* a4 = (const v4f*)a; const v4f* b4 = (const v4f*)b; v4f* y4 = (v4f*)y;
  for (long i = (long)blockIdx.x * blockDim.x + threadIdx.x; i < n4;
       i += (long)gridDim.x * blockDim.x) {
    v4f s = a4[i] + b4[i];
#pragma unroll
    for (int e = 0; e < 4; ++e) s[e] = fmaxf(s[e], 0.f);
    y4[i] = s;
  }
}

// top-M by descending score; slot = exact rank (ties by lower index) ->
// deterministic, reproduces jax.lax.top_k order with no atomics.
__global__ void topm_kernel(const float* __restrict__ s, int* __restrict__ sel,
                            int N, int M) {
  long i = (long)blockIdx.x * blockDim.x + threadIdx.x;
  if (i >= (long)BATCH * N) return;
  int b = (int)(i / N); int k = (int)(i % N);
  const float* sb = s + (long)b * N;
  float sk = sb[k];
  int r = 0;
  for (int j = 0; j < N; ++j) {
    float sj = sb[j];
    r += (sj > sk) || (sj == sk && j < k);
  }
  if (r < M) sel[(long)b * M + r] = k;
}

__global__ void gather_pos_kernel(const float* __restrict__ pos, const int* __restrict__ sel,
                                  float* __restrict__ out, int N, int M) {
  long total = (long)BATCH * M * 3;
  for (long i = (long)blockIdx.x * blockDim.x + threadIdx.x; i < total;
       i += (long)gridDim.x * blockDim.x) {
    int c = (int)(i % 3); long t = i / 3;
    int m = (int)(t % M); int b = (int)(t / M);
    out[i] = pos[((long)b * N + sel[(long)b * M + m]) * 3 + c];
  }
}

__global__ void maxpool16_kernel(const float* __restrict__ f, float* __restrict__ out,
                                 long rows, int C) {
  const int C4 = C >> 2;
  const v4f* f4 = (const v4f*)f; v4f* o4 = (v4f*)out;
  long total = rows * C4;
  for (long i = (long)blockIdx.x * blockDim.x + threadIdx.x; i < total;
       i += (long)gridDim.x * blockDim.x) {
    int c4 = (int)(i % C4); long r = i / C4;
    const v4f* p = f4 + (r * 16) * (long)C4 + c4;
    v4f m = p[0];
#pragma unroll
    for (int j = 1; j < 16; ++j) {
      v4f pv = p[(long)j * C4];
#pragma unroll
      for (int e = 0; e < 4; ++e) m[e] = fmaxf(m[e], pv[e]);
    }
    o4[i] = m;
  }
}

__global__ void mean_kernel(const float* __restrict__ x, float* __restrict__ out,
                            int N, int C) {
  long i = (long)blockIdx.x * blockDim.x + threadIdx.x;
  if (i >= (long)BATCH * C) return;
  int c = (int)(i % C); int b = (int)(i / C);
  float s = 0.f;
  for (int n = 0; n < N; ++n) s += x[((long)b * N + n) * (long)C + c];
  out[i] = s / N;
}

// ============================ host-side plumbing ============================
struct BlockP {
  const float *b1, *b2, *b3, *g1, *g2, *g3, *w1, *w3;
  const float *ab1, *ab2, *abe, *ag, *aw1, *aw2;
  const float *pb1, *pb2, *pbe, *pg, *pw1, *pw2;
  const float *wk, *wq, *wv;
};
struct TdP {
  const float *b1, *b2, *b3, *be1, *be2, *g1, *g2, *w1, *w2, *w3;
};
struct WS {
  float *F0, *F1, *F2, *F3, *F4, *F5;
  float *NKK, *NV, *NPE, *NT1, *NT2;
  int *IDX; float *SCORES; int *SEL;
  float *POS1, *POS2, *MEANB, *HEADB;
};

static BlockP getBlock(void* const* d, int base) {
  BlockP P;
  P.b1  = (const float*)d[base + 0];  P.b2  = (const float*)d[base + 1];
  P.b3  = (const float*)d[base + 2];  P.g1  = (const float*)d[base + 3];
  P.g2  = (const float*)d[base + 4];  P.g3  = (const float*)d[base + 5];
  P.ab1 = (const float*)d[base + 6];  P.ab2 = (const float*)d[base + 7];
  P.abe = (const float*)d[base + 8];  P.ag  = (const float*)d[base + 9];
  P.aw1 = (const float*)d[base + 10]; P.aw2 = (const float*)d[base + 11];
  P.pb1 = (const float*)d[base + 12]; P.pb2 = (const float*)d[base + 13];
  P.pbe = (const float*)d[base + 14]; P.pg  = (const float*)d[base + 15];
  P.pw1 = (const float*)d[base + 16]; P.pw2 = (const float*)d[base + 17];
  P.wk  = (const float*)d[base + 18]; P.wq  = (const float*)d[base + 19];
  P.wv  = (const float*)d[base + 20];
  P.w1  = (const float*)d[base + 21]; P.w3  = (const float*)d[base + 22];
  return P;
}
static TdP getTd(void* const* d, int base) {
  TdP T;
  T.b1 = (const float*)d[base + 0]; T.b2 = (const float*)d[base + 1];
  T.b3 = (const float*)d[base + 2]; T.be1 = (const float*)d[base + 3];
  T.be2 = (const float*)d[base + 4]; T.g1 = (const float*)d[base + 5];
  T.g2 = (const float*)d[base + 6]; T.w1 = (const float*)d[base + 7];
  T.w2 = (const float*)d[base + 8]; T.w3 = (const float*)d[base + 9];
  return T;
}

static inline unsigned eltBlocks(long total) {
  long b = (total + 255) / 256;
  if (b < 1) b = 1;
  if (b > 1048576) b = 1048576;
  return (unsigned)b;
}

static void gemm(const float* X, const float* W, const float* bias, float* Y,
                 long rows, int Cin, int Cout, int relu, hipStream_t s) {
  long tiles = ((rows + 15) / 16) * (long)((Cout + 15) / 16);
  long blocks = (tiles + 7) / 8;
  gemm_wmma_kernel<<<dim3((unsigned)blocks), dim3(256), 0, s>>>(X, W, bias, Y,
                                                                rows, Cin, Cout, relu);
}
static void ln(const float* X, const float* g, const float* b, float* Y,
               long rows, int C, int relu, hipStream_t s) {
  long blocks = (rows * 32 + 255) / 256;
  ln_kernel<<<dim3((unsigned)blocks), dim3(256), 0, s>>>(X, g, b, Y, rows, C, relu);
}
static void knn(const float* qp, const float* rp, int* idx, int Mq, int Nr,
                hipStream_t s) {
  int bpb = Mq / 128;
  knn_kernel<<<dim3((unsigned)(BATCH * bpb)), dim3(128), 0, s>>>(qp, rp, idx, Mq, Nr, bpb);
}

static void run_block(const float* X, float* Xout, const float* pos, const BlockP& P,
                      int N, int d, WS& w, hipStream_t s) {
  long BN = (long)BATCH * N;
  long nb4 = BN * 16 * (d / 4);
  gemm(X, P.w1, nullptr, w.F2, BN, d, d, 0, s);
  ln(w.F2, P.g1, P.b1, w.F2, BN, d, 1, s);                        // net
  knn(pos, pos, w.IDX, N, N, s);
  gemm(w.F2, P.wq, nullptr, w.F3, BN, d, d, 0, s);                // q
  gemm(w.F2, P.wk, nullptr, w.F4, BN, d, d, 0, s);
  gather_feat_kernel<<<eltBlocks(nb4), 256, 0, s>>>(w.F4, w.IDX, w.NKK, N, N, d);
  gemm(w.F2, P.wv, nullptr, w.F4, BN, d, d, 0, s);
  gather_feat_kernel<<<eltBlocks(nb4), 256, 0, s>>>(w.F4, w.IDX, w.NV, N, N, d);
  rel_kernel<<<eltBlocks(BN * 16 * 3), 256, 0, s>>>(pos, w.IDX, w.NT1, N);
  gemm(w.NT1, P.pw1, P.pb1, w.NT2, BN * 16, 3, 12, 0, s);
  ln(w.NT2, P.pg, P.pbe, w.NT2, BN * 16, 12, 1, s);
  gemm(w.NT2, P.pw2, P.pb2, w.NPE, BN * 16, 12, d, 0, s);         // pe
  wcombine_kernel<<<eltBlocks(nb4), 256, 0, s>>>(w.F3, w.NKK, w.NPE, w.NT1, N, d);
  gemm(w.NT1, P.aw1, P.ab1, w.NT2, BN * 16, d, d, 0, s);
  ln(w.NT2, P.ag, P.abe, w.NT2, BN * 16, d, 1, s);
  gemm(w.NT2, P.aw2, P.ab2, w.NT1, BN * 16, d, d, 0, s);          // attn logits
  attn_kernel<<<eltBlocks(BN * (d / 4)), 256, 0, s>>>(w.NT1, w.NV, w.NPE, w.F4, N, d);
  ln(w.F4, P.g2, P.b2, w.F4, BN, d, 1, s);
  gemm(w.F4, P.w3, nullptr, w.F5, BN, d, d, 0, s);
  ln(w.F5, P.g3, P.b3, w.F5, BN, d, 0, s);
  relu_add_kernel<<<eltBlocks(BN * d / 4), 256, 0, s>>>(w.F5, X, Xout, BN * d / 4);
}

static void run_td(const float* X, float* Xout, const float* posIn, float* posOut,
                   const TdP& T, int N, int M, int ci, int co, WS& w, hipStream_t s) {
  long BN = (long)BATCH * N;
  gemm(X, T.w1, T.b1, w.F2, BN, ci, co, 0, s);
  ln(w.F2, T.g1, T.be1, w.F2, BN, co, 1, s);
  gemm(w.F2, T.w2, T.b2, w.F3, BN, co, co, 0, s);
  ln(w.F3, T.g2, T.be2, w.F3, BN, co, 1, s);                      // f
  gemm(w.F3, T.w3, T.b3, w.SCORES, BN, co, 1, 0, s);
  topm_kernel<<<dim3((unsigned)((BATCH * (long)N + 255) / 256)), dim3(256), 0, s>>>(
      w.SCORES, w.SEL, N, M);
  gather_pos_kernel<<<eltBlocks((long)BATCH * M * 3), 256, 0, s>>>(posIn, w.SEL, posOut, N, M);
  knn(posOut, posIn, w.IDX, M, N, s);
  gather_feat_kernel<<<eltBlocks((long)BATCH * M * 16 * (co / 4)), 256, 0, s>>>(
      w.F3, w.IDX, w.NT1, N, M, co);
  maxpool16_kernel<<<eltBlocks((long)BATCH * M * (co / 4)), 256, 0, s>>>(
      w.NT1, Xout, (long)BATCH * M, co);
}

// Input flattening assumption: JAX pytree order (dict keys sorted, recursive):
// 0: points; 1..23 block0 {b1,b2,b3,g1,g2,g3, tr{ab1,ab2,abe,ag,aw1,aw2,pb1,pb2,
// pbe,pg,pw1,pw2,wk,wq,wv}, w1,w3}; 24..46 block1; 47..69 block2; 70..73 fc1b,
// fc1w,fc2b,fc2w; 74..77 hb1,hb2,hw1,hw2; 78..87 td1{b1,b2,b3,be1,be2,g1,g2,w1,
// w2,w3}; 88..97 td2.
extern "C" void kernel_launch(void* const* d_in, const int* in_sizes, int n_in,
                              void* d_out, int out_size, void* d_ws, size_t ws_size,
                              hipStream_t stream) {
  (void)in_sizes; (void)out_size;
  if (n_in < 98) return;

  const float* points = (const float*)d_in[0];
  BlockP B0 = getBlock(d_in, 1);
  BlockP B1 = getBlock(d_in, 24);
  BlockP B2 = getBlock(d_in, 47);
  const float* fc1b = (const float*)d_in[70];
  const float* fc1w = (const float*)d_in[71];
  const float* fc2b = (const float*)d_in[72];
  const float* fc2w = (const float*)d_in[73];
  const float* hb1  = (const float*)d_in[74];
  const float* hb2  = (const float*)d_in[75];
  const float* hw1  = (const float*)d_in[76];
  const float* hw2  = (const float*)d_in[77];
  TdP T1 = getTd(d_in, 78);
  TdP T2 = getTd(d_in, 88);

  // -------- workspace layout (floats) --------
  const size_t FBSZ = 2097152;    // 4*8192*64  (max feature buffer)
  const size_t NBSZ = 16777216;   // 4*8192*16*32 (max neighbor tensor)
  float* ws = (float*)d_ws;
  size_t o = 0;
  WS w;
  w.F0 = ws + o; o += FBSZ;  w.F1 = ws + o; o += FBSZ;
  w.F2 = ws + o; o += FBSZ;  w.F3 = ws + o; o += FBSZ;
  w.F4 = ws + o; o += FBSZ;  w.F5 = ws + o; o += FBSZ;
  w.NKK = ws + o; o += NBSZ; w.NV  = ws + o; o += NBSZ;
  w.NPE = ws + o; o += NBSZ; w.NT1 = ws + o; o += NBSZ;
  w.NT2 = ws + o; o += NBSZ;
  w.IDX = (int*)(ws + o); o += 524288;       // 4*8192*16
  w.SCORES = ws + o; o += 32768;             // 4*8192
  w.SEL = (int*)(ws + o); o += 8192;         // 4*2048
  w.POS1 = ws + o; o += 24576;               // 4*2048*3
  w.POS2 = ws + o; o += 6144;                // 4*512*3
  w.MEANB = ws + o; o += 512;                // 4*128
  w.HEADB = ws + o; o += 256;                // 4*64
  if (ws_size < o * sizeof(float)) return;

  // -------- forward pass --------
  const long N0 = 8192;
  gemm(points, fc1w, fc1b, w.F0, BATCH * N0, 3, 32, 1, stream);
  gemm(w.F0, fc2w, fc2b, w.F1, BATCH * N0, 32, 32, 0, stream);

  run_block(w.F1, w.F0, points, B0, 8192, 32, w, stream);
  run_td(w.F0, w.F1, points, w.POS1, T1, 8192, 2048, 32, 64, w, stream);
  run_block(w.F1, w.F0, w.POS1, B1, 2048, 64, w, stream);
  run_td(w.F0, w.F1, w.POS1, w.POS2, T2, 2048, 512, 64, 128, w, stream);
  run_block(w.F1, w.F0, w.POS2, B2, 512, 128, w, stream);

  mean_kernel<<<dim3((unsigned)((BATCH * 128 + 255) / 256)), dim3(256), 0, stream>>>(
      w.F0, w.MEANB, 512, 128);
  gemm(w.MEANB, hw1, hb1, w.HEADB, 4, 128, 64, 1, stream);
  gemm(w.HEADB, hw2, hb2, (float*)d_out, 4, 64, 40, 0, stream);
}
